// Imaginator_89300960018772
// MI455X (gfx1250) — matrix operand order repaired and checked by reading, not compiled, not run on hardware
//
#include <hip/hip_runtime.h>

typedef __attribute__((ext_vector_type(16))) _Float16 v16h;
typedef __attribute__((ext_vector_type(8)))  _Float16 v8h;
typedef __attribute__((ext_vector_type(8)))  float    v8f;

namespace {

constexpr int BATCH   = 8192;
constexpr int NPLANET = 64;
constexpr int BPW     = 4;    // batch rows per workgroup
constexpr int NTILES  = 16;   // 16-planet row tiles per workgroup (8 waves x 2)

// ---------------- LDS layout (bytes) ----------------
// Weight tiles: each 32x16 (KxN) f16 tile = 512 halves = 1024 B,
// lane-major: lane n owns 32 contiguous bytes; element e of lane n is
// B[kc*32 + (n>=16?8:0) + e + (e>=8?8:0)][jt*16 + (n&15)].
// Row k == Ksrc of each weight matrix holds the BIAS vector; the matching
// activation column is forced to 1.0, so C accumulators start from zero.
constexpr int W0_TILES = 1 * 10;   // K:32  (src 10 +bias row) x N:160 (src 150)
constexpr int W1_TILES = 5 * 10;   // K:160 (src 150+bias row) x N:160 (src 150)
constexpr int W4_TILES = 5 * 7;    // K:160 (src 150+bias row) x N:112 (src 100)

constexpr int W0_OFF = 0;
constexpr int W1_OFF = W0_OFF + W0_TILES * 1024;            // 10240
constexpr int W2_OFF = W1_OFF + W1_TILES * 1024;            // 61440
constexpr int W3_OFF = W2_OFF + W1_TILES * 1024;            // 112640
constexpr int W4_OFF = W3_OFF + W1_TILES * 1024;            // 163840
constexpr int STG_OFF = W4_OFF + W4_TILES * 1024;           // 199680

constexpr int STG_STRIDE     = 176;                         // halves/row (rows 16B aligned)
constexpr int STG_TILE_BYTES = 16 * STG_STRIDE * 2;         // 5632 B per row tile
constexpr int PART_OFF = STG_OFF + NTILES * STG_TILE_BYTES; // 289792
constexpr int PART_SZ  = NTILES * 2 * 112 * 4;              // 14336 (tile x lane-half slots)
constexpr int AGG_OFF  = PART_OFF + PART_SZ;                // 304128
constexpr int AGG_SZ   = BPW * 112 * 4;                     // 1792
constexpr int H1_OFF   = AGG_OFF + AGG_SZ;                  // 305920
constexpr int H1_SZ    = BPW * 100 * 4;                     // 1600
constexpr int SMEM_BYTES = H1_OFF + H1_SZ;                  // 307520 < 320KB

__device__ inline v16h make_v16h(v8h lo, v8h hi) {
  v16h r;
#pragma unroll
  for (int e = 0; e < 8; ++e) { r[e] = lo[e]; r[e + 8] = hi[e]; }
  return r;
}

// Load one 32x16 B tile from swizzled LDS (2 x 16B aligned loads per lane).
__device__ inline v16h load_btile(const _Float16* wl, int kc, int jt, int njt, int lane) {
  const _Float16* tp = wl + (((kc * njt) + jt) << 9) + (lane << 4);
  return make_v16h(*(const v8h*)tp, *(const v8h*)(tp + 8));
}

// Reload activations from row-major f16 staging into WMMA A-layout.
// A 16x32 f16 layout: lane L holds row L&15; elements 0..7 = K {base..base+7},
// elements 8..15 = K {16+base..16+base+7}, base = (L<16 ? 0 : 8).
__device__ inline void load_a(v16h* a, const _Float16* stg, int lane) {
  const _Float16* srow = stg + (lane & 15) * STG_STRIDE;
  const int base = (lane < 16) ? 0 : 8;
#pragma unroll
  for (int kc = 0; kc < 5; ++kc) {
    v8h lo = *(const v8h*)(srow + kc * 32 + base);
    v8h hi = *(const v8h*)(srow + kc * 32 + 16 + base);
    a[kc] = make_v16h(lo, hi);
  }
}

// Cooperative f32->f16 weight+bias swizzle into LDS B-tile layout (zero padded).
// Weight row k==ksrc carries the bias vector.
__device__ inline void fill_weights(_Float16* dst, const float* __restrict__ src,
                                    const float* __restrict__ bias,
                                    int ksrc, int nsrc, int kcN, int njt, int tid) {
  const int total = kcN * njt * 512;   // halves
  for (int o = tid; o < total; o += 256) {
    const int tile = o >> 9;
    const int rem  = o & 511;
    const int lane = rem >> 4;
    const int e    = rem & 15;
    const int kc = tile / njt;
    const int jt = tile % njt;
    const int k = kc * 32 + ((lane >> 4) << 3) + e + ((e >> 3) << 3);
    const int c = jt * 16 + (lane & 15);
    float v = 0.f;
    if (c < nsrc) {
      if (k < ksrc)       v = src[k * nsrc + c];
      else if (k == ksrc) v = bias[c];
    }
    dst[o] = (_Float16)v;
  }
}

// Build layer-0 A fragment for row tile T: row = planet, K = concat(agent, planet),
// K=10 set to 1.0 (bias row of W0).
__device__ inline v16h build_in(const float* __restrict__ agent,
                                const float* __restrict__ planets,
                                int b0, int T, int lane) {
  const int bb = b0 + (T >> 2);
  const int p  = (T & 3) * 16 + (lane & 15);
  const float* ag = agent + bb * 5;
  const float* pl = planets + (bb * NPLANET + p) * 5;
  float f[16];
#pragma unroll
  for (int i = 0; i < 16; ++i) f[i] = 0.f;
  f[0] = ag[0]; f[1] = ag[1]; f[2] = ag[2]; f[3] = ag[3]; f[4] = ag[4];
  f[5] = pl[0]; f[6] = pl[1]; f[7] = pl[2]; f[8] = pl[3]; f[9] = pl[4];
  f[10] = 1.f;                                // bias marker (K == Ksrc)
  const int kb = (lane < 16) ? 0 : 8;
  v16h a;
#pragma unroll
  for (int e = 0; e < 8; ++e) a[e] = (_Float16)f[kb + e];
#pragma unroll
  for (int e = 8; e < 16; ++e) a[e] = (_Float16)0.f;   // K >= 16 padding
  return a;
}

// One dense layer for TWO row tiles sharing each software-pipelined B tile:
// D = relu(A @ W) in f16 domain, staged row-major; column mark_col forced to 1.0
// (bias marker for the next layer's bias row).
template <int KC, int NJT>
__device__ inline void run_layer2(const _Float16* wl, const v16h* a0, const v16h* a1,
                                  _Float16* stg0, _Float16* stg1, int lane,
                                  int mark_col) {
  const int cin   = lane & 15;
  const int rbase = (lane < 16) ? 0 : 8;
  const v8h zero = {};
  v16h bm = load_btile(wl, 0, 0, NJT, lane);
#pragma unroll
  for (int jt = 0; jt < NJT; ++jt) {
    const int col = jt * 16 + cin;
    v8f c0 = {}, c1 = {};
#pragma unroll
    for (int kc = 0; kc < KC; ++kc) {
      int nk = kc + 1, nj = jt;
      if (nk == KC) { nk = 0; ++nj; }
      v16h bnext = bm;
      if (nj < NJT) bnext = load_btile(wl, nk, nj, NJT, lane);  // preload next tile
      c0 = __builtin_amdgcn_wmma_f32_16x16x32_f16(false, a0[kc], false, bm,
                                                  (short)0, c0, false, false);
      c1 = __builtin_amdgcn_wmma_f32_16x16x32_f16(false, a1[kc], false, bm,
                                                  (short)0, c1, false, false);
      bm = bnext;
    }
    v8h h0 = __builtin_elementwise_max(__builtin_convertvector(c0, v8h), zero);
    v8h h1 = __builtin_elementwise_max(__builtin_convertvector(c1, v8h), zero);
    if ((mark_col >> 4) == jt && col == mark_col) {   // outer test constant-folds
#pragma unroll
      for (int r = 0; r < 8; ++r) { h0[r] = (_Float16)1.f; h1[r] = (_Float16)1.f; }
    }
#pragma unroll
    for (int r = 0; r < 8; ++r) {
      stg0[(rbase + r) * STG_STRIDE + col] = h0[r];
      stg1[(rbase + r) * STG_STRIDE + col] = h1[r];
    }
  }
}

// Final relation layer (two tiles): relu then per-lane f32 sum of the 8 rows in C.
template <int KC, int NJT>
__device__ inline void run_layer2_sum(const _Float16* wl, const v16h* a0, const v16h* a1,
                                      float* part0, float* part1, int lane) {
  const int cin = lane & 15;
  v16h bm = load_btile(wl, 0, 0, NJT, lane);
#pragma unroll
  for (int jt = 0; jt < NJT; ++jt) {
    const int col = jt * 16 + cin;
    v8f c0 = {}, c1 = {};
#pragma unroll
    for (int kc = 0; kc < KC; ++kc) {
      int nk = kc + 1, nj = jt;
      if (nk == KC) { nk = 0; ++nj; }
      v16h bnext = bm;
      if (nj < NJT) bnext = load_btile(wl, nk, nj, NJT, lane);
      c0 = __builtin_amdgcn_wmma_f32_16x16x32_f16(false, a0[kc], false, bm,
                                                  (short)0, c0, false, false);
      c1 = __builtin_amdgcn_wmma_f32_16x16x32_f16(false, a1[kc], false, bm,
                                                  (short)0, c1, false, false);
      bm = bnext;
    }
    float s0 = 0.f, s1 = 0.f;
#pragma unroll
    for (int r = 0; r < 8; ++r) { s0 += fmaxf(c0[r], 0.f); s1 += fmaxf(c1[r], 0.f); }
    part0[col] = s0;   // partial sum over 8 planets for this column
    part1[col] = s1;
  }
}

__global__ __launch_bounds__(256)
void relnet_kernel(const float* __restrict__ agent, const float* __restrict__ planets,
                   const float* __restrict__ action,
                   const float* __restrict__ rw0, const float* __restrict__ rb0,
                   const float* __restrict__ rw1, const float* __restrict__ rb1,
                   const float* __restrict__ rw2, const float* __restrict__ rb2,
                   const float* __restrict__ rw3, const float* __restrict__ rb3,
                   const float* __restrict__ rw4, const float* __restrict__ rb4,
                   const float* __restrict__ ow0, const float* __restrict__ ob0,
                   const float* __restrict__ ow1, const float* __restrict__ ob1,
                   float* __restrict__ out) {
  extern __shared__ char smem[];
  _Float16* w0 = (_Float16*)(smem + W0_OFF);
  _Float16* w1 = (_Float16*)(smem + W1_OFF);
  _Float16* w2 = (_Float16*)(smem + W2_OFF);
  _Float16* w3 = (_Float16*)(smem + W3_OFF);
  _Float16* w4 = (_Float16*)(smem + W4_OFF);
  float* part = (float*)(smem + PART_OFF);
  float* agg  = (float*)(smem + AGG_OFF);
  float* h1   = (float*)(smem + H1_OFF);

  const int tid  = threadIdx.x;
  const int lane = tid & 31;
  const int w    = tid >> 5;            // wave 0..7
  const int b0   = blockIdx.x * BPW;    // 4 batch rows per workgroup

  // Warm L2 for the big weight matrices (global_prefetch_b8).
  {
    const int off = tid * 64;           // floats (256 B apart per thread)
    if (off < 150 * 150) {
      __builtin_prefetch(rw1 + off, 0, 1);
      __builtin_prefetch(rw2 + off, 0, 1);
      __builtin_prefetch(rw3 + off, 0, 1);
    }
    if (off < 150 * 100) __builtin_prefetch(rw4 + off, 0, 1);
  }

  // ---- one-time cooperative weight(+bias) swizzle into LDS ----
  fill_weights(w0, rw0, rb0, 10, 150, 1, 10, tid);
  fill_weights(w1, rw1, rb1, 150, 150, 5, 10, tid);
  fill_weights(w2, rw2, rb2, 150, 150, 5, 10, tid);
  fill_weights(w3, rw3, rb3, 150, 150, 5, 10, tid);
  fill_weights(w4, rw4, rb4, 150, 100, 5, 7, tid);
  __syncthreads();

  // ---- relation MLP: each wave owns TWO 16-planet row tiles ----
  const int T0 = w * 2;
  const int T1 = w * 2 + 1;
  _Float16* stg0 = (_Float16*)(smem + STG_OFF + T0 * STG_TILE_BYTES);
  _Float16* stg1 = (_Float16*)(smem + STG_OFF + T1 * STG_TILE_BYTES);

  v16h a0[5], a1[5];
  a0[0] = build_in(agent, planets, b0, T0, lane);
  a1[0] = build_in(agent, planets, b0, T1, lane);

  run_layer2<1, 10>(w0, a0, a1, stg0, stg1, lane, 150);
  load_a(a0, stg0, lane); load_a(a1, stg1, lane);
  run_layer2<5, 10>(w1, a0, a1, stg0, stg1, lane, 150);
  load_a(a0, stg0, lane); load_a(a1, stg1, lane);
  run_layer2<5, 10>(w2, a0, a1, stg0, stg1, lane, 150);
  load_a(a0, stg0, lane); load_a(a1, stg1, lane);
  run_layer2<5, 10>(w3, a0, a1, stg0, stg1, lane, 150);
  load_a(a0, stg0, lane); load_a(a1, stg1, lane);
  run_layer2_sum<5, 7>(w4, a0, a1,
                       part + (T0 * 2 + (lane >> 4)) * 112,
                       part + (T1 * 2 + (lane >> 4)) * 112, lane);
  __syncthreads();

  // ---- deterministic cross-wave reduction over planets ----
  // agg[bq][c] = sum over 4 tiles x 2 lane-half slots, fixed order.
  for (int idx = tid; idx < BPW * 112; idx += 256) {
    const int bq = idx / 112, c = idx % 112;
    float s = 0.f;
#pragma unroll
    for (int tt = 0; tt < 4; ++tt)
#pragma unroll
      for (int h = 0; h < 2; ++h)
        s += part[((bq * 4 + tt) * 2 + h) * 112 + c];
    agg[bq * 112 + c] = s;
  }
  __syncthreads();

  // ---- object MLP (fp32; 0.2% of FLOPs) ----
  for (int idx = tid; idx < BPW * 100; idx += 256) {
    const int ob = idx / 100, j = idx % 100;
    const float* ag = agent + (b0 + ob) * 5;
    const float* ac = action + (b0 + ob) * 2;
    float acc = ob0[j];
#pragma unroll
    for (int i = 0; i < 5; ++i) acc += ag[i] * ow0[i * 100 + j];
#pragma unroll
    for (int i = 0; i < 2; ++i) acc += ac[i] * ow0[(5 + i) * 100 + j];
    const float* ga = agg + ob * 112;
    for (int i = 0; i < 100; ++i) acc += ga[i] * ow0[(7 + i) * 100 + j];
    h1[ob * 100 + j] = fmaxf(acc, 0.f);
  }
  __syncthreads();
  if (tid < BPW * 4) {
    const int ob = tid >> 2, o = tid & 3;
    float acc = ob1[o];
    const float* hh = h1 + ob * 100;
    for (int j = 0; j < 100; ++j) acc += hh[j] * ow1[j * 4 + o];
    out[(b0 + ob) * 4 + o] = acc;
  }
}

}  // namespace

extern "C" void kernel_launch(void* const* d_in, const int* in_sizes, int n_in,
                              void* d_out, int out_size, void* d_ws, size_t ws_size,
                              hipStream_t stream) {
  (void)in_sizes; (void)n_in; (void)out_size; (void)d_ws; (void)ws_size;
  const float* agent   = (const float*)d_in[0];
  const float* planets = (const float*)d_in[1];
  const float* action  = (const float*)d_in[2];
  const float* rw0 = (const float*)d_in[3];  const float* rb0 = (const float*)d_in[4];
  const float* rw1 = (const float*)d_in[5];  const float* rb1 = (const float*)d_in[6];
  const float* rw2 = (const float*)d_in[7];  const float* rb2 = (const float*)d_in[8];
  const float* rw3 = (const float*)d_in[9];  const float* rb3 = (const float*)d_in[10];
  const float* rw4 = (const float*)d_in[11]; const float* rb4 = (const float*)d_in[12];
  const float* ow0 = (const float*)d_in[13]; const float* ob0 = (const float*)d_in[14];
  const float* ow1 = (const float*)d_in[15]; const float* ob1 = (const float*)d_in[16];
  float* out = (float*)d_out;

  // Allow >64KB dynamic LDS (gfx1250 supports up to 320KB per workgroup).
  (void)hipFuncSetAttribute((const void*)relnet_kernel,
                            hipFuncAttributeMaxDynamicSharedMemorySize, SMEM_BYTES);

  relnet_kernel<<<BATCH / BPW, 256, SMEM_BYTES, stream>>>(
      agent, planets, action,
      rw0, rb0, rw1, rb1, rw2, rb2, rw3, rb3, rw4, rb4,
      ow0, ob0, ow1, ob1, out);
}
